// MultiHeadSelfAttention_7301444403685
// MI455X (gfx1250) — matrix operand order repaired and checked by reading, not compiled, not run on hardware
//
#include <hip/hip_runtime.h>

typedef __attribute__((ext_vector_type(16))) _Float16 v16h;
typedef __attribute__((ext_vector_type(8)))  _Float16 v8h;
typedef __attribute__((ext_vector_type(8)))  float    v8f;

#define SEQ_L   2048
#define NHEAD   16
#define DHEAD   64

#if defined(__has_builtin)
#if __has_builtin(__builtin_amdgcn_tensor_load_to_lds) && \
    __has_builtin(__builtin_amdgcn_s_wait_tensorcnt)
#define USE_TDM 1
#endif
#endif
#ifndef USE_TDM
#define USE_TDM 0
#endif

__device__ __forceinline__ v16h cat8(v8h lo, v8h hi) {
  return __builtin_shufflevector(lo, hi, 0,1,2,3,4,5,6,7,8,9,10,11,12,13,14,15);
}

__device__ __forceinline__ v8f wmma_f16(v16h a, v16h b, v8f c) {
  // D = A(16x32 f16) * B(32x16 f16) + C(16x16 f32)
  return __builtin_amdgcn_wmma_f32_16x16x32_f16(
      /*neg_a=*/false, a, /*neg_b=*/false, b,
      /*c_mod=*/(short)0, c, /*reuse_a=*/false, /*reuse_b=*/false);
}

#if USE_TDM
typedef __attribute__((ext_vector_type(4))) unsigned int u32x4;
typedef __attribute__((ext_vector_type(8))) int          i32x8;
typedef __attribute__((ext_vector_type(4))) int          i32x4;

// Build a 2D Tensor-DMA descriptor (D#) and issue TENSOR_LOAD_TO_LDS.
//   dsz_code: 0=1B,1=2B,2=4B   tile_d0: contiguous elems/row   tile_d1: rows
//   stride0:  row stride in elements
__device__ __forceinline__ void tdm_load_tile(
    unsigned lds_addr, const void* gptr, unsigned dsz_code,
    unsigned tile_d0, unsigned tile_d1,
    unsigned tensor_d0, unsigned tensor_d1,
    unsigned long long stride0)
{
  const unsigned long long ga = (unsigned long long)(size_t)gptr;
  u32x4 g0;
  g0[0] = 1u;                                             // count=1, user D#
  g0[1] = lds_addr;                                       // lds_addr [63:32]
  g0[2] = (unsigned)ga;                                   // global_addr lo
  g0[3] = (unsigned)((ga >> 32) & 0x1FFFFFFull)           // global_addr hi (57b)
        | (2u << 30);                                     // type=2 ("image")
  i32x8 g1;
  g1[0] = (int)(dsz_code << 16);                          // data_size [17:16]
  g1[1] = (int)((tensor_d0 & 0xFFFFu) << 16);             // tensor_dim0 [79:48]
  g1[2] = (int)((tensor_d0 >> 16) | ((tensor_d1 & 0xFFFFu) << 16));
  g1[3] = (int)((tensor_d1 >> 16) | (tile_d0 << 16));     // tile_dim0 [127:112]
  g1[4] = (int)(tile_d1 & 0xFFFFu);                       // tile_dim1, tile_dim2=0
  g1[5] = (int)(stride0 & 0xFFFFFFFFull);                 // dim0_stride [207:160]
  g1[6] = (int)((stride0 >> 32) & 0xFFFFull);
  g1[7] = 0;
  const i32x4 z4 = {0, 0, 0, 0};
  const i32x8 z8 = {0, 0, 0, 0, 0, 0, 0, 0};
  __builtin_amdgcn_tensor_load_to_lds(g0, g1, z4, z4, z8, 0);
}
#endif

// ---------------------------------------------------------------------------
// Projection GEMM: Out[bh][l][d] (f16) = X[4096,1024] @ W[1024,1024] (+bias)
// Block = 128 threads (4 waves), 64(M) x 64(N) tile; weight tile DMA'd by TDM
// into a double buffer, converted/transposed to f16 in LDS.
// ---------------------------------------------------------------------------
__global__ __launch_bounds__(128) void proj_kernel(
    const float* __restrict__ X,
    const float* __restrict__ W,
    const float* __restrict__ bias,
    _Float16* __restrict__ Out,
    int has_bias)
{
#if USE_TDM
  __shared__ float Wraw[2][32 * 64];        // raw f32 weight tiles (TDM dest)
#endif
  __shared__ _Float16 Bt[64 * 32];          // weight tile, transposed [n][k], f16
  const int tid   = threadIdx.x;
  const int wave  = tid >> 5;
  const int lane  = tid & 31;
  const int lr    = lane & 15;              // row (A) / col (B,C)
  const int kgrpA = (lane >> 4) * 8;        // A-frag K group base
  const int kbB   = (lane >> 4) * 16;       // unused; B-frag also uses kgrpA
  const int m0    = blockIdx.x * 64;
  const int n0    = blockIdx.y * 64;

  v8f acc[4] = {};
  const float* arow = X + (size_t)(m0 + wave * 16 + lr) * 1024;

#if USE_TDM
  if (wave == 0)
    tdm_load_tile((unsigned)(size_t)&Wraw[0][0], W + n0,
                  /*f32*/2, /*tile*/64, 32, /*tensor*/64, 1024, /*stride*/1024);
#endif

  for (int k0 = 0, kt = 0; k0 < 1024; k0 += 32, ++kt) {
#if USE_TDM
    const int cur = kt & 1, nxt = cur ^ 1;
    if (wave == 0) __builtin_amdgcn_s_wait_tensorcnt((short)0);
    __syncthreads();                        // Wraw[cur] ready; Bt reads done
    #pragma unroll
    for (int i = 0; i < 4; ++i) {           // LDS f32 -> LDS f16 transposed
      const int e  = tid * 16 + i * 4;
      const int kk = e >> 6;
      const int nn = e & 63;
      const float4 wv = *(const float4*)&Wraw[cur][kk * 64 + nn];
      Bt[(nn + 0) * 32 + kk] = (_Float16)wv.x;
      Bt[(nn + 1) * 32 + kk] = (_Float16)wv.y;
      Bt[(nn + 2) * 32 + kk] = (_Float16)wv.z;
      Bt[(nn + 3) * 32 + kk] = (_Float16)wv.w;
    }
    if (k0 + 32 < 1024 && wave == 0)        // overlap next tile DMA with WMMA
      tdm_load_tile((unsigned)(size_t)&Wraw[nxt][0],
                    W + (size_t)(k0 + 32) * 1024 + n0,
                    2, 64, 32, 64, 1024, 1024);
    __syncthreads();                        // Bt ready
#else
    __syncthreads();
    #pragma unroll
    for (int i = 0; i < 4; ++i) {
      const int e  = tid * 16 + i * 4;
      const int kk = e >> 6;
      const int nn = e & 63;
      const float4 wv = *(const float4*)(W + (size_t)(k0 + kk) * 1024 + n0 + nn);
      Bt[(nn + 0) * 32 + kk] = (_Float16)wv.x;
      Bt[(nn + 1) * 32 + kk] = (_Float16)wv.y;
      Bt[(nn + 2) * 32 + kk] = (_Float16)wv.z;
      Bt[(nn + 3) * 32 + kk] = (_Float16)wv.w;
    }
    __syncthreads();
#endif

    // A fragment: row = lr, K = kgrpA..+7 and 16+kgrpA..+7 (f32 -> f16)
    const float4 a0 = *(const float4*)(arow + k0 + kgrpA);
    const float4 a1 = *(const float4*)(arow + k0 + kgrpA + 4);
    const float4 a2 = *(const float4*)(arow + k0 + 16 + kgrpA);
    const float4 a3 = *(const float4*)(arow + k0 + 16 + kgrpA + 4);
    v16h af;
    af[0]  = (_Float16)a0.x; af[1]  = (_Float16)a0.y;
    af[2]  = (_Float16)a0.z; af[3]  = (_Float16)a0.w;
    af[4]  = (_Float16)a1.x; af[5]  = (_Float16)a1.y;
    af[6]  = (_Float16)a1.z; af[7]  = (_Float16)a1.w;
    af[8]  = (_Float16)a2.x; af[9]  = (_Float16)a2.y;
    af[10] = (_Float16)a2.z; af[11] = (_Float16)a2.w;
    af[12] = (_Float16)a3.x; af[13] = (_Float16)a3.y;
    af[14] = (_Float16)a3.z; af[15] = (_Float16)a3.w;

    #pragma unroll
    for (int f = 0; f < 4; ++f) {
      const v16h bf = cat8(*(const v8h*)(&Bt[(f * 16 + lr) * 32 + kgrpA]),
                           *(const v8h*)(&Bt[(f * 16 + lr) * 32 + 16 + kgrpA]));
      acc[f] = wmma_f16(af, bf, acc[f]);
    }
  }

  // epilogue: C/D layout row = v + 8*(lane>>4), col = lr
  const int rb = 8 * (lane >> 4);
  #pragma unroll
  for (int f = 0; f < 4; ++f) {
    const int n  = n0 + f * 16 + lr;
    const float bv = has_bias ? bias[n] : 0.0f;
    const int h = n >> 6;
    const int d = n & 63;
    #pragma unroll
    for (int v = 0; v < 8; ++v) {
      const int m    = m0 + wave * 16 + v + rb;
      const int bidx = m >> 11;            // / SEQ_L
      const int lseq = m & (SEQ_L - 1);
      Out[(((size_t)(bidx * NHEAD + h)) * SEQ_L + lseq) * DHEAD + d] =
          (_Float16)(acc[f][v] + bv);
    }
  }
}

// ---------------------------------------------------------------------------
// Flash attention: one (b,h, 64-q-row tile) per 128-thread block.
// V tiles are DMA'd by TDM into a double buffer and transposed in LDS.
// ---------------------------------------------------------------------------
__global__ __launch_bounds__(128) void attn_kernel(
    const _Float16* __restrict__ qbuf,   // [B*H][L][64]
    const _Float16* __restrict__ kbuf,
    const _Float16* __restrict__ vbuf,
    const int* __restrict__ mask,        // [B][L]
    float* __restrict__ out)             // [B][L][H*64]
{
#if USE_TDM
  __shared__ _Float16 Vraw[2][64 * 64];  // raw V tiles [key][d] (TDM dest)
#endif
  __shared__ _Float16 Vt[64 * 64];       // V tile transposed [d][key]
  __shared__ _Float16 Plds[4 * 16 * 64]; // per-wave P staging
  const int tid   = threadIdx.x;
  const int wave  = tid >> 5;
  const int lane  = tid & 31;
  const int qt    = blockIdx.x;
  const int bh    = blockIdx.y;
  const int b     = bh >> 4;
  const int h     = bh & 15;
  const int mb    = bh & 1;              // reference quirk: (b*H+h) % B
  const int lr    = lane & 15;
  const int kgrpA = (lane >> 4) * 8;
  const int kbB   = (lane >> 4) * 16;

  // Q A-fragments (d = 0..31 and 32..63), held in registers for whole kernel
  const _Float16* qrow =
      qbuf + (((size_t)bh * SEQ_L) + qt * 64 + wave * 16 + lr) * DHEAD;
  const v16h qf0 = cat8(*(const v8h*)(qrow + kbB),
                        *(const v8h*)(qrow + kbB + 8));
  const v16h qf1 = cat8(*(const v8h*)(qrow + 32 + kbB),
                        *(const v8h*)(qrow + 40 + kbB));

  float mi[8], li[8];
  #pragma unroll
  for (int v = 0; v < 8; ++v) { mi[v] = -1e30f; li[v] = 0.0f; }
  v8f o[4] = {};

  const _Float16* kb_head = kbuf + (size_t)bh * SEQ_L * DHEAD;
  const _Float16* vb_head = vbuf + (size_t)bh * SEQ_L * DHEAD;
  _Float16* Pw = Plds + wave * (16 * 64);
  const int rb = 8 * (lane >> 4);

#if USE_TDM
  if (wave == 0)
    tdm_load_tile((unsigned)(size_t)&Vraw[0][0], vb_head,
                  /*f16*/1, /*tile*/64, 64, /*tensor*/64, SEQ_L, /*stride*/64);
#endif

  for (int kt = 0; kt < SEQ_L / 64; ++kt) {
    const int key0 = kt * 64;
#if USE_TDM
    const int cur = kt & 1, nxt = cur ^ 1;
    if (wave == 0) __builtin_amdgcn_s_wait_tensorcnt((short)0);
    __syncthreads();                     // Vraw[cur] ready; old Vt reads done
    #pragma unroll
    for (int i = 0; i < 4; ++i) {        // LDS -> LDS transpose
      const int e   = tid * 32 + i * 8;
      const int key = e >> 6;
      const int d   = e & 63;
      const v8h vv = *(const v8h*)&Vraw[cur][key * 64 + d];
      #pragma unroll
      for (int j = 0; j < 8; ++j) Vt[(d + j) * 64 + key] = vv[j];
    }
    if (kt + 1 < SEQ_L / 64 && wave == 0)  // overlap next V tile DMA
      tdm_load_tile((unsigned)(size_t)&Vraw[nxt][0],
                    vb_head + (size_t)(key0 + 64) * DHEAD,
                    1, 64, 64, 64, SEQ_L, 64);
    __syncthreads();                     // Vt ready
#else
    __syncthreads();
    #pragma unroll
    for (int i = 0; i < 4; ++i) {
      const int e   = tid * 32 + i * 8;
      const int key = e >> 6;
      const int d   = e & 63;
      const v8h vv = *(const v8h*)(vb_head + (size_t)(key0 + key) * DHEAD + d);
      #pragma unroll
      for (int j = 0; j < 8; ++j) Vt[(d + j) * 64 + key] = vv[j];
    }
    __syncthreads();
#endif

    // S = Q @ K^T over 4 key sub-tiles; scale + mask on accumulators
    v8f sc[4];
    float pr[4][8];
    #pragma unroll
    for (int f = 0; f < 4; ++f) {
      const _Float16* krow = kb_head + (size_t)(key0 + f * 16 + lr) * DHEAD;
      const v16h kf0 = *(const v16h*)(krow + kbB);        // d = kbB..kbB+15
      const v16h kf1 = *(const v16h*)(krow + 32 + kbB);   // d = 32+kbB..
      v8f s = {};
      s = wmma_f16(qf0, kf0, s);
      s = wmma_f16(qf1, kf1, s);
      const int mv = mask[mb * SEQ_L + key0 + f * 16 + lr];
      #pragma unroll
      for (int v = 0; v < 8; ++v) {
        const float x = s[v] * 0.125f;    // 1/sqrt(64)
        s[v] = (mv == 0) ? -1e9f : x;
      }
      sc[f] = s;
    }

    // online softmax per accumulator row (row = v + rb, spread over 16 lanes)
    #pragma unroll
    for (int v = 0; v < 8; ++v) {
      float rmax = fmaxf(fmaxf(sc[0][v], sc[1][v]), fmaxf(sc[2][v], sc[3][v]));
      #pragma unroll
      for (int off = 8; off > 0; off >>= 1)
        rmax = fmaxf(rmax, __shfl_xor(rmax, off, 32));
      const float mnew  = fmaxf(mi[v], rmax);
      const float alpha = __expf(mi[v] - mnew);
      float rsum = 0.0f;
      #pragma unroll
      for (int f = 0; f < 4; ++f) {
        const float p = __expf(sc[f][v] - mnew);
        pr[f][v] = p;
        rsum += p;
      }
      #pragma unroll
      for (int off = 8; off > 0; off >>= 1)
        rsum += __shfl_xor(rsum, off, 32);
      li[v] = li[v] * alpha + rsum;
      mi[v] = mnew;
      #pragma unroll
      for (int f = 0; f < 4; ++f) o[f][v] *= alpha;
    }

    // P (C/D layout) -> LDS -> A-fragment layout
    #pragma unroll
    for (int f = 0; f < 4; ++f)
      #pragma unroll
      for (int v = 0; v < 8; ++v)
        Pw[(v + rb) * 64 + f * 16 + lr] = (_Float16)pr[f][v];
    __syncthreads();

    const _Float16* prow = Pw + lr * 64;
    const v16h pf0 = cat8(*(const v8h*)(prow + kbB),
                          *(const v8h*)(prow + kbB + 8));
    const v16h pf1 = cat8(*(const v8h*)(prow + 32 + kbB),
                          *(const v8h*)(prow + 40 + kbB));

    // O += P @ V   (B-fragments from transposed V in LDS)
    #pragma unroll
    for (int nf = 0; nf < 4; ++nf) {
      const _Float16* vcol = Vt + (nf * 16 + lr) * 64;
      const v16h vf0 = *(const v16h*)(vcol + kbB);
      const v16h vf1 = *(const v16h*)(vcol + 32 + kbB);
      o[nf] = wmma_f16(pf0, vf0, o[nf]);
      o[nf] = wmma_f16(pf1, vf1, o[nf]);
    }

    // prefetch next K tile into cache hierarchy
    if (kt + 1 < SEQ_L / 64)
      __builtin_prefetch(kb_head + (size_t)(key0 + 64 + lane) * DHEAD, 0, 1);
  }

  // epilogue: normalize and store f32 out[b][q][h*64+d]
  #pragma unroll
  for (int nf = 0; nf < 4; ++nf) {
    const int d = nf * 16 + lr;
    #pragma unroll
    for (int v = 0; v < 8; ++v) {
      const int q = qt * 64 + wave * 16 + v + rb;
      out[((size_t)(b * SEQ_L + q)) * (NHEAD * DHEAD) + h * DHEAD + d] =
          o[nf][v] / li[v];
    }
  }
}

// ---------------------------------------------------------------------------
extern "C" void kernel_launch(void* const* d_in, const int* in_sizes, int n_in,
                              void* d_out, int out_size, void* d_ws, size_t ws_size,
                              hipStream_t stream) {
  const float* Q    = (const float*)d_in[0];
  const float* K    = (const float*)d_in[1];
  const float* V    = (const float*)d_in[2];
  const int*   mask = (const int*)d_in[3];
  const float* W_Q  = (const float*)d_in[4];
  const float* b_Q  = (const float*)d_in[5];
  const float* W_K  = (const float*)d_in[6];
  const float* W_V  = (const float*)d_in[7];
  const float* b_V  = (const float*)d_in[8];
  float* out = (float*)d_out;

  const size_t proj_elems = (size_t)2 * SEQ_L * NHEAD * DHEAD; // 4096*1024
  _Float16* qbuf = (_Float16*)d_ws;
  _Float16* kbuf = qbuf + proj_elems;
  _Float16* vbuf = kbuf + proj_elems;

  dim3 pgrid(64, 16);   // 4096/64 M-tiles x 1024/64 N-tiles
  proj_kernel<<<pgrid, 128, 0, stream>>>(Q, W_Q, b_Q, qbuf, 1);
  proj_kernel<<<pgrid, 128, 0, stream>>>(K, W_K, nullptr, kbuf, 0);
  proj_kernel<<<pgrid, 128, 0, stream>>>(V, W_V, b_V, vbuf, 1);

  dim3 agrid(SEQ_L / 64, 2 * NHEAD); // 32 q-tiles x 32 (b,h)
  attn_kernel<<<agrid, 128, 0, stream>>>(qbuf, kbuf, vbuf, mask, out);
}
